// Block_65214783422695
// MI455X (gfx1250) — compile-verified, matrix operand order checked by reading, not run on hardware
//
#include <hip/hip_runtime.h>
#include <math.h>

typedef _Float16 half_t;
typedef __attribute__((ext_vector_type(16))) _Float16 v16h;
typedef __attribute__((ext_vector_type(8)))  float    v8f;

#define BB 16
#define TT 512
#define CC 464
#define HH 16
#define DD 29
#define CP 480            // C padded to multiple of 32 (K dim)
#define MM (BB*TT)        // 8192 token rows
#define NQKV 1536         // 3 * H * 32 (D padded to 32)
#define NPROJ 512         // 464 padded to 64-multiple
#define NW1 1856          // 4*C, already 64-multiple
#define NW2 512           // 464 padded
#define KW2 1856
#define EPSF 1e-5f

// ---- fragment loader: CDNA5 16-bit A-matrix 16x32 layout ----
// lane 0-15: row=lane, VGPR v holds K = (v>>2)*16 + (v&3)*2 (+1)
// lane 16-31: same rows, K offset +8.  B operands are staged transposed
// ([N][K]) so the identical loader produces the B fragment.
__device__ __forceinline__ v16h load_frag(const half_t* p, int ld) {
  int lane = threadIdx.x & 31;
  const half_t* r = p + (size_t)(lane & 15) * ld + ((lane >> 4) << 3);
  v16h f;
#pragma unroll
  for (int v = 0; v < 8; ++v) {
    int k = ((v >> 2) << 4) + ((v & 3) << 1);
    f[2 * v]     = r[k];
    f[2 * v + 1] = r[k + 1];
  }
  return f;
}

__device__ __forceinline__ v8f wmma16(v16h a, v16h b, v8f c) {
  return __builtin_amdgcn_wmma_f32_16x16x32_f16(false, a, false, b,
                                                (short)0, c, false, false);
}

// ---- 32x64 wave macro-tile: 2 A-frags share each B-frag (8 WMMA / 12 b128) ----
template <int KDIM>
__device__ __forceinline__ void gemm_tile_32x64(const half_t* __restrict__ A,
                                                const half_t* __restrict__ Bt,
                                                int m0, int n0, v8f acc[2][4]) {
  const half_t* arow0 = A + (size_t)m0 * KDIM;
  const half_t* arow1 = A + (size_t)(m0 + 16) * KDIM;
  for (int k0 = 0; k0 < KDIM; k0 += 32) {
    if (k0 + 32 < KDIM) {                 // lowers to global_prefetch_b8
      __builtin_prefetch(arow0 + k0 + 32, 0, 1);
      __builtin_prefetch(arow1 + k0 + 32, 0, 1);
      __builtin_prefetch(Bt + (size_t)n0 * KDIM + k0 + 32, 0, 1);
    }
    v16h a0 = load_frag(arow0 + k0, KDIM);
    v16h a1 = load_frag(arow1 + k0, KDIM);
#pragma unroll
    for (int t = 0; t < 4; ++t) {
      v16h b = load_frag(Bt + (size_t)(n0 + 16 * t) * KDIM + k0, KDIM);
      acc[0][t] = wmma16(a0, b, acc[0][t]);
      acc[1][t] = wmma16(a1, b, acc[1][t]);
    }
  }
}

// ---- weight packing ----
__global__ void pack_qkv_kernel(const float* __restrict__ wq,
                                const float* __restrict__ wk,
                                const float* __restrict__ wv,
                                half_t* __restrict__ dst) {
  int idx = blockIdx.x * 256 + threadIdx.x;
  if (idx >= NQKV * CP) return;
  int n = idx / CP, k = idx % CP;
  int mat = n >> 9, h = (n >> 5) & 15, d = n & 31;
  float v = 0.f;
  if (d < DD && k < CC) {
    const float* w = (mat == 0) ? wq : (mat == 1) ? wk : wv;
    v = w[((size_t)h * CC + k) * DD + d];
  }
  dst[idx] = (half_t)v;
}

__global__ void pack_T_kernel(const float* __restrict__ src, half_t* __restrict__ dst,
                              int K, int N, int Kpad, int Npad) {
  int idx = blockIdx.x * 256 + threadIdx.x;
  if (idx >= Npad * Kpad) return;
  int n = idx / Kpad, k = idx % Kpad;
  dst[idx] = (k < K && n < N) ? (half_t)src[(size_t)k * N + n] : (half_t)0.f;
}

__global__ void zero_h_kernel(half_t* p, int n) {
  int idx = blockIdx.x * 256 + threadIdx.x;
  if (idx < n) p[idx] = (half_t)0.f;
}

// ---- RMSNorm (f32 in -> f16 padded out) ----
__global__ __launch_bounds__(256) void rmsnorm_kernel(const float* __restrict__ x,
                                                      const float* __restrict__ g,
                                                      half_t* __restrict__ dst) {
  __shared__ float red[8];
  int row = blockIdx.x;
  const float* xr = x + (size_t)row * CC;
  float p = 0.f;
  for (int c = threadIdx.x; c < CC; c += 256) { float v = xr[c]; p += v * v; }
#pragma unroll
  for (int m = 16; m >= 1; m >>= 1) p += __shfl_xor(p, m, 32);
  if ((threadIdx.x & 31) == 0) red[threadIdx.x >> 5] = p;
  __syncthreads();
  if (threadIdx.x == 0) {
    float s = 0.f;
#pragma unroll
    for (int i = 0; i < 8; ++i) s += red[i];
    red[0] = s;
  }
  __syncthreads();
  float r = rsqrtf(red[0] / (float)CC + EPSF);
  half_t* drow = dst + (size_t)row * CP;
  for (int c = threadIdx.x; c < CP; c += 256)
    drow[c] = (c < CC) ? (half_t)(xr[c] * r * g[c]) : (half_t)0.f;
}

// ---- QKV GEMM: [8192 x 480] x [480 x 1536] with scatter epilogue ----
__global__ __launch_bounds__(256) void gemm_qkv_kernel(
    const half_t* __restrict__ A, const half_t* __restrict__ Bt,
    half_t* __restrict__ qf, half_t* __restrict__ kf, half_t* __restrict__ vT) {
  const int nT4 = NQKV / 64;
  int tile = blockIdx.x * 8 + (threadIdx.x >> 5);
  int m0 = (tile / nT4) * 32;
  int n0 = (tile % nT4) * 64;
  v8f acc[2][4] = {};
  gemm_tile_32x64<CP>(A, Bt, m0, n0, acc);
  int lane = threadIdx.x & 31;
  int nl = lane & 15, mh = (lane >> 4) * 8;
#pragma unroll
  for (int i = 0; i < 2; ++i) {
#pragma unroll
    for (int t = 0; t < 4; ++t) {
      int n = n0 + 16 * t + nl;
      int mat = n >> 9, h = (n >> 5) & 15, d = n & 31;
#pragma unroll
      for (int j = 0; j < 8; ++j) {
        int m = m0 + 16 * i + mh + j;
        int b_ = m >> 9, tok = m & 511;
        half_t val = (half_t)acc[i][t][j];
        if (mat == 0)      qf[((size_t)(b_ * HH + h) * TT + tok) * 32 + d] = val;
        else if (mat == 1) kf[((size_t)(b_ * HH + h) * TT + tok) * 32 + d] = val;
        else               vT[((size_t)(b_ * HH + h) * 32 + d) * TT + tok] = val;
      }
    }
  }
}

// ---- fused attention per (b, h, 16-token tile) ----
__global__ __launch_bounds__(256) void attn_kernel(
    const half_t* __restrict__ qf, const half_t* __restrict__ kf,
    const half_t* __restrict__ vT, const float* __restrict__ pos,
    half_t* __restrict__ o16) {
  __shared__ float  sc[16 * 512];
  __shared__ half_t pf[16 * 512];
  int tTile = blockIdx.x, h = blockIdx.y, b = blockIdx.z;
  int t0 = tTile * 16;
  int wave = threadIdx.x >> 5, lane = threadIdx.x & 31;
  int nl = lane & 15, mh = (lane >> 4) * 8;
  const float scale = 0.18569533817705186f;  // 29^-0.5
  size_t bh = (size_t)(b * HH + h);
  int kpad = ((tTile + 2) >> 1) * 32;        // valid s-range rounded to 32

  // scores: Q(16x32) * K^T, bias + causal mask fused
  v16h aq = load_frag(qf + (bh * TT + t0) * 32, 32);
  for (int st = wave; st <= tTile; st += 8) {
    v16h bk = load_frag(kf + (bh * TT + st * 16) * 32, 32);
    v8f sf = {};
    sf = wmma16(aq, bk, sf);
#pragma unroll
    for (int j = 0; j < 8; ++j) {
      int m = mh + j, t = t0 + m, s = st * 16 + nl;
      float val = (s <= t) ? (sf[j] + pos[h * TT + (t - s)]) * scale
                           : -__builtin_inff();
      sc[m * 512 + s] = val;
    }
  }
  __syncthreads();

  // row softmax: 16 threads per row, shfl_xor reductions within 16 lanes
  {
    int r = threadIdx.x >> 4, c = threadIdx.x & 15;
    int nv = t0 + r + 1;
    float mx = -__builtin_inff();
    for (int s = c; s < nv; s += 16) mx = fmaxf(mx, sc[r * 512 + s]);
#pragma unroll
    for (int m = 8; m >= 1; m >>= 1) mx = fmaxf(mx, __shfl_xor(mx, m, 16));
    float sum = 0.f;
    for (int s = c; s < nv; s += 16) sum += __expf(sc[r * 512 + s] - mx);
#pragma unroll
    for (int m = 8; m >= 1; m >>= 1) sum += __shfl_xor(sum, m, 16);
    float inv = 1.0f / sum;
    for (int s = c; s < kpad; s += 16)
      pf[r * 512 + s] = (s < nv) ? (half_t)(__expf(sc[r * 512 + s] - mx) * inv)
                                 : (half_t)0.f;
  }
  __syncthreads();

  // O = P(16 x kpad) * V(kpad x 32): waves 0/1 take the two 16-wide n-tiles
  if (wave < 2) {
    int n0 = wave * 16;
    v8f acc = {};
    for (int k0 = 0; k0 < kpad; k0 += 32) {
      v16h a  = load_frag(&pf[k0], 512);
      v16h bv = load_frag(vT + (bh * 32 + n0) * TT + k0, TT);
      acc = wmma16(a, bv, acc);
    }
    int d = n0 + nl;
    if (d < DD) {
#pragma unroll
      for (int j = 0; j < 8; ++j) {
        int t = t0 + mh + j;
        o16[((size_t)b * TT + t) * CP + h * DD + d] = (half_t)acc[j];
      }
    }
  }
}

// ---- proj GEMM + residual + bias ----
__global__ __launch_bounds__(256) void gemm_proj_kernel(
    const half_t* __restrict__ A, const half_t* __restrict__ Bt,
    const float* __restrict__ x, const float* __restrict__ bias,
    float* __restrict__ out) {
  const int nT4 = NPROJ / 64;
  int tile = blockIdx.x * 8 + (threadIdx.x >> 5);
  int m0 = (tile / nT4) * 32;
  int n0 = (tile % nT4) * 64;
  v8f acc[2][4] = {};
  gemm_tile_32x64<CP>(A, Bt, m0, n0, acc);
  int lane = threadIdx.x & 31;
  int nl = lane & 15, mh = (lane >> 4) * 8;
#pragma unroll
  for (int i = 0; i < 2; ++i) {
#pragma unroll
    for (int t = 0; t < 4; ++t) {
      int n = n0 + 16 * t + nl;
      if (n >= CC) continue;
#pragma unroll
      for (int j = 0; j < 8; ++j) {
        int m = m0 + 16 * i + mh + j;
        out[(size_t)m * CC + n] = x[(size_t)m * CC + n] + acc[i][t][j] + bias[n];
      }
    }
  }
}

// ---- MLP GEMM1 + SiLU -> f16 ----
__global__ __launch_bounds__(256) void gemm_mlp1_kernel(
    const half_t* __restrict__ A, const half_t* __restrict__ Bt,
    half_t* __restrict__ a3) {
  const int nT4 = NW1 / 64;
  int tile = blockIdx.x * 8 + (threadIdx.x >> 5);
  int m0 = (tile / nT4) * 32;
  int n0 = (tile % nT4) * 64;
  v8f acc[2][4] = {};
  gemm_tile_32x64<CP>(A, Bt, m0, n0, acc);
  int lane = threadIdx.x & 31;
  int nl = lane & 15, mh = (lane >> 4) * 8;
#pragma unroll
  for (int i = 0; i < 2; ++i) {
#pragma unroll
    for (int t = 0; t < 4; ++t) {
      int n = n0 + 16 * t + nl;
#pragma unroll
      for (int j = 0; j < 8; ++j) {
        int m = m0 + 16 * i + mh + j;
        float v = acc[i][t][j];
        a3[(size_t)m * NW1 + n] = (half_t)(v / (1.f + __expf(-v)));
      }
    }
  }
}

// ---- MLP GEMM2 + residual -> d_out ----
__global__ __launch_bounds__(256) void gemm_mlp2_kernel(
    const half_t* __restrict__ A, const half_t* __restrict__ Bt,
    const float* __restrict__ x2, float* __restrict__ out) {
  const int nT4 = NW2 / 64;
  int tile = blockIdx.x * 8 + (threadIdx.x >> 5);
  int m0 = (tile / nT4) * 32;
  int n0 = (tile % nT4) * 64;
  v8f acc[2][4] = {};
  gemm_tile_32x64<KW2>(A, Bt, m0, n0, acc);
  int lane = threadIdx.x & 31;
  int nl = lane & 15, mh = (lane >> 4) * 8;
#pragma unroll
  for (int i = 0; i < 2; ++i) {
#pragma unroll
    for (int t = 0; t < 4; ++t) {
      int n = n0 + 16 * t + nl;
      if (n >= CC) continue;
#pragma unroll
      for (int j = 0; j < 8; ++j) {
        int m = m0 + 16 * i + mh + j;
        out[(size_t)m * CC + n] = x2[(size_t)m * CC + n] + acc[i][t][j];
      }
    }
  }
}

extern "C" void kernel_launch(void* const* d_in, const int* in_sizes, int n_in,
                              void* d_out, int out_size, void* d_ws, size_t ws_size,
                              hipStream_t stream) {
  (void)in_sizes; (void)n_in; (void)out_size; (void)ws_size;
  const float* x     = (const float*)d_in[0];
  const float* wq    = (const float*)d_in[2];
  const float* wk    = (const float*)d_in[3];
  const float* wv    = (const float*)d_in[4];
  const float* pos   = (const float*)d_in[5];
  const float* wproj = (const float*)d_in[6];
  const float* bproj = (const float*)d_in[7];
  const float* g1    = (const float*)d_in[8];
  const float* g2    = (const float*)d_in[9];
  const float* w1    = (const float*)d_in[10];
  const float* w2    = (const float*)d_in[11];
  float* out = (float*)d_out;

  char* ws = (char*)d_ws;
  size_t off = 0;
  auto alloc = [&](size_t bytes) -> char* {
    char* p = ws + off;
    off += (bytes + 255) & ~(size_t)255;
    return p;
  };
  half_t* h1    = (half_t*)alloc((size_t)MM * CP * 2);
  half_t* Bqkv  = (half_t*)alloc((size_t)NQKV * CP * 2);
  half_t* Bproj = (half_t*)alloc((size_t)NPROJ * CP * 2);
  half_t* Bw1   = (half_t*)alloc((size_t)NW1 * CP * 2);
  half_t* Bw2   = (half_t*)alloc((size_t)NW2 * KW2 * 2);
  half_t* qf    = (half_t*)alloc((size_t)BB * HH * TT * 32 * 2);
  half_t* kf    = (half_t*)alloc((size_t)BB * HH * TT * 32 * 2);
  half_t* vT    = (half_t*)alloc((size_t)BB * HH * TT * 32 * 2);
  half_t* o16   = (half_t*)alloc((size_t)MM * CP * 2);
  float*  x2    = (float*) alloc((size_t)MM * CC * 4);
  half_t* h2    = (half_t*)alloc((size_t)MM * CP * 2);
  half_t* a3    = (half_t*)alloc((size_t)MM * NW1 * 2);

  pack_qkv_kernel<<<(NQKV * CP + 255) / 256, 256, 0, stream>>>(wq, wk, wv, Bqkv);
  pack_T_kernel<<<(NPROJ * CP + 255) / 256, 256, 0, stream>>>(wproj, Bproj, CC, CC, CP, NPROJ);
  pack_T_kernel<<<(NW1 * CP + 255) / 256, 256, 0, stream>>>(w1, Bw1, CC, NW1, CP, NW1);
  pack_T_kernel<<<(NW2 * KW2 + 255) / 256, 256, 0, stream>>>(w2, Bw2, KW2, CC, KW2, NW2);
  zero_h_kernel<<<(MM * CP + 255) / 256, 256, 0, stream>>>(o16, MM * CP);

  rmsnorm_kernel<<<MM, 256, 0, stream>>>(x, g1, h1);
  gemm_qkv_kernel<<<(MM / 32) * (NQKV / 64) / 8, 256, 0, stream>>>(h1, Bqkv, qf, kf, vT);

  dim3 ag(TT / 16, HH, BB);
  attn_kernel<<<ag, 256, 0, stream>>>(qf, kf, vT, pos, o16);

  gemm_proj_kernel<<<(MM / 32) * (NPROJ / 64) / 8, 256, 0, stream>>>(o16, Bproj, x, bproj, x2);
  rmsnorm_kernel<<<MM, 256, 0, stream>>>(x2, g2, h2);
  gemm_mlp1_kernel<<<(MM / 32) * (NW1 / 64) / 8, 256, 0, stream>>>(h2, Bw1, a3);
  gemm_mlp2_kernel<<<(MM / 32) * (NW2 / 64) / 8, 256, 0, stream>>>(a3, Bw2, x2, out);
}
